// GatedGraphNeuralNetwork_40896678593053
// MI455X (gfx1250) — compile-verified, matrix operand order checked by reading, not run on hardware
//
#include <hip/hip_runtime.h>

// Problem constants (match reference).
#define NN 50000
#define HH 128
#define TT 4
#define EE 150000
#define EPB 80            // edges per msg block; EE % EPB == 0, EPB % 16 == 0
#define ETILES (EPB / 16)

typedef __attribute__((ext_vector_type(16))) __bf16 v16bf;
typedef __attribute__((ext_vector_type(8)))  float  v8f;

union BF16Frag { v16bf v; uint4 q[2]; };

// ---------------------------------------------------------------------------
// gfx1250 async global->LDS staging (ASYNCcnt-tracked). The builtin takes
// typed pointers: (v4i addrspace(1)*, v4i addrspace(3)*, imm offset, imm cpol).
// Guarded so the file still compiles on toolchains without the builtins.
// ---------------------------------------------------------------------------
#if __has_builtin(__builtin_amdgcn_global_load_async_to_lds_b128) && \
    __has_builtin(__builtin_amdgcn_s_wait_asynccnt)
#define USE_ASYNC_LDS 1
#else
#define USE_ASYNC_LDS 0
#endif

typedef int v4i __attribute__((vector_size(16)));
typedef __attribute__((address_space(1))) v4i* gbl_v4i_ptr;
typedef __attribute__((address_space(3))) v4i* lds_v4i_ptr;

__device__ __forceinline__ void copy16_to_lds(void* lds, const void* g) {
#if USE_ASYNC_LDS
    // Flat LDS addresses carry the raw LDS byte offset in the low 32 bits
    // (aperture rules, ISA 10.2); integer casts sidestep addrspacecast rules.
    __builtin_amdgcn_global_load_async_to_lds_b128(
        (gbl_v4i_ptr)(unsigned long long)(uintptr_t)g,
        (lds_v4i_ptr)(unsigned)(uintptr_t)lds,
        0, 0);
#else
    *(uint4*)lds = *(const uint4*)g;
#endif
}

__device__ __forceinline__ void async_join() {
#if USE_ASYNC_LDS
    __builtin_amdgcn_s_wait_asynccnt(0);
#endif
}

// ---------------------------------------------------------------------------
// WMMA helpers
// ---------------------------------------------------------------------------

// Load one WMMA 16-bit A/B fragment (16 bf16 per lane) following the CDNA5
// 16-bit layout: caller folds base = kk + (lane>=16 ? 8 : 0) into p0; this
// lane then needs K = +0..+7 and K = +16..+23 -> two 16B loads.
__device__ __forceinline__ v16bf load_frag(const unsigned short* p0) {
    BF16Frag f;
    const uint4* p = (const uint4*)p0;
    f.q[0] = p[0];
    f.q[1] = p[2];
    return f.v;
}

__device__ __forceinline__ v8f wmma_bf16(v16bf a, v16bf b, v8f c) {
    return __builtin_amdgcn_wmma_f32_16x16x32_bf16(false, a, false, b,
                                                   (short)0, c, false, false);
}

__device__ __forceinline__ unsigned short f2bf(float x) {
    union { float f; unsigned u; } v; v.f = x;
    unsigned r = v.u + 0x7fffu + ((v.u >> 16) & 1u);   // round-to-nearest-even
    return (unsigned short)(r >> 16);
}

__device__ __forceinline__ float sigmoidf_(float x) {
    return 1.0f / (1.0f + __expf(-x));
}

// ---------------------------------------------------------------------------
// Elementwise utility kernels
// ---------------------------------------------------------------------------
__global__ __launch_bounds__(256) void zero_f32(float* p, int n) {
    int i = blockIdx.x * 256 + threadIdx.x;
    if (i < n) p[i] = 0.0f;
}

__global__ __launch_bounds__(256) void cvt_bf16(const float* __restrict__ in,
                                                unsigned short* __restrict__ out, int n) {
    int i = blockIdx.x * 256 + threadIdx.x;
    if (i < n) out[i] = f2bf(in[i]);
}

// out[r*ostride + ooff + c] = bf16(in[r*cols + c])
__global__ __launch_bounds__(256) void cvt_bf16_strided(const float* __restrict__ in,
                                                        unsigned short* __restrict__ out,
                                                        int cols, int ostride, int ooff, int n) {
    int i = blockIdx.x * 256 + threadIdx.x;
    if (i < n) {
        int r = i / cols, c = i - r * cols;
        out[(long)r * ostride + ooff + c] = f2bf(in[i]);
    }
}

// ---------------------------------------------------------------------------
// Message passing: block = EPB edges (grid.x) x one edge-type (grid.y).
// 32KB weight tile staged once per block, amortized over EPB edges; each wave
// holds its 4 B-fragments in registers across ETILES edge sub-tiles. Scatter
// via f32 atomics (SGPR base + pre-scaled 32-bit byte voffsets).
// ---------------------------------------------------------------------------
__global__ __launch_bounds__(256) void msg_kernel(
    const unsigned short* __restrict__ hb,     // N x 128 bf16
    const int*            __restrict__ edges,  // T x E x 2
    const unsigned short* __restrict__ Wb,     // T x 128 x 128 bf16 (this layer)
    const float*          __restrict__ bias,   // T x 128 (this layer)
    float*                __restrict__ inc)    // N x 128 f32
{
    __shared__ __align__(16) unsigned short As[EPB * HH];  // gathered h rows (20KB)
    __shared__ __align__(16) unsigned short Ws[HH * HH];   // W tile (32KB)
    __shared__ unsigned tgtoff[EPB];                       // tgt * row-bytes

    const int t   = blockIdx.y;
    const int e0  = blockIdx.x * EPB;
    const int tid = threadIdx.x;
    const int* ep = edges + ((long)t * EE + e0) * 2;

    if (tid < EPB) tgtoff[tid] = (unsigned)ep[tid * 2 + 1] * (HH * 4u);
    // gather EPB src rows: 16 x 16B chunks per row
    for (int i = tid; i < EPB * 16; i += 256) {
        int row = i >> 4, ch = i & 15;
        int s = ep[row * 2];
        copy16_to_lds((uint4*)As + i, (const uint4*)(hb + (long)s * HH) + ch);
    }
    {   // stage 32KB weight tile: 2048 x 16B / 256 threads
        const uint4* wg = (const uint4*)(Wb + (long)t * HH * HH);
        for (int i = tid; i < HH * HH / 8; i += 256)
            copy16_to_lds((uint4*)Ws + i, wg + i);
    }
    async_join();
    __syncthreads();

    const int wave = tid >> 5, lane = tid & 31;
    const int half = lane >> 4, idx = lane & 15;
    const int n = wave * 16 + idx;               // output feature column
    const unsigned n4 = (unsigned)n * 4u;        // column byte offset
    char* const incb = (char*)inc;

    // B fragments: loaded once, reused for all edge sub-tiles.
    const unsigned short* wrow = Ws + n * HH + half * 8;
    const v16bf B0 = load_frag(wrow + 0);
    const v16bf B1 = load_frag(wrow + 32);
    const v16bf B2 = load_frag(wrow + 64);
    const v16bf B3 = load_frag(wrow + 96);
    const float bv = bias[t * HH + n];

    #pragma unroll
    for (int s = 0; s < ETILES; ++s) {
        const unsigned short* arow = As + (s * 16 + idx) * HH + half * 8;
        v8f c = {};
        c = wmma_bf16(load_frag(arow + 0),  B0, c);
        c = wmma_bf16(load_frag(arow + 32), B1, c);
        c = wmma_bf16(load_frag(arow + 64), B2, c);
        c = wmma_bf16(load_frag(arow + 96), B3, c);
        #pragma unroll
        for (int v = 0; v < 8; ++v) {
            int m = s * 16 + v + half * 8;       // edge row in block
            atomicAdd((float*)(incb + (tgtoff[m] + n4)), c[v] + bv);
        }
    }
}

// ---------------------------------------------------------------------------
// Fused GRU cell: block = 32 nodes (two 16-node sub-tiles sharing each
// B-fragment -> halves L2 weight traffic). Wave w owns output columns
// c = 16w+idx and computes the r/z/n gate tiles for both gx and gh, so the
// gating runs entirely in registers. Also emits the bf16 image of h_new
// (fusing the per-timestep f32->bf16 conversion). 32-bit byte offsets.
// ---------------------------------------------------------------------------
template <int KIN>
__global__ __launch_bounds__(256) void gru_kernel(
    const unsigned short* __restrict__ xinb,  // N x KIN bf16
    const unsigned short* __restrict__ hb,    // N x 128 bf16 (read; rewritten)
    const float*          __restrict__ hold,  // N x 128 f32
    const unsigned short* __restrict__ Wih,   // 384 x KIN bf16
    const unsigned short* __restrict__ Whh,   // 384 x 128 bf16
    const float*          __restrict__ bih,   // 384
    const float*          __restrict__ bhh,   // 384
    float*                __restrict__ hout,  // N x 128 f32
    unsigned short*       __restrict__ hbout) // N x 128 bf16 (next-step state)
{
    __shared__ __align__(16) unsigned short Xs[32 * KIN];
    __shared__ __align__(16) unsigned short Hs[32 * HH];

    const int node0 = blockIdx.x * 32;
    const int tid = threadIdx.x;
    for (int i = tid; i < 32 * KIN / 8; i += 256) {
        int row = i / (KIN / 8), ch = i % (KIN / 8);
        int node = node0 + row; if (node > NN - 1) node = NN - 1;   // clamp tail
        copy16_to_lds((uint4*)Xs + i, (const uint4*)(xinb + (size_t)node * KIN) + ch);
    }
    for (int i = tid; i < 32 * HH / 8; i += 256) {
        int row = i >> 4, ch = i & 15;
        int node = node0 + row; if (node > NN - 1) node = NN - 1;
        copy16_to_lds((uint4*)Hs + i, (const uint4*)(hb + (size_t)node * HH) + ch);
    }
    async_join();
    __syncthreads();

    const int wave = tid >> 5, lane = tid & 31;
    const int half = lane >> 4, idx = lane & 15;
    const int c = wave * 16 + idx;               // output h column [0,128)

    v8f axr0 = {}, axz0 = {}, axn0 = {}, ahr0 = {}, ahz0 = {}, ahn0 = {};
    v8f axr1 = {}, axz1 = {}, axn1 = {}, ahr1 = {}, ahz1 = {}, ahn1 = {};

    #pragma unroll
    for (int kk = 0; kk < KIN; kk += 32) {
        v16bf a0 = load_frag(Xs + idx * KIN + kk + half * 8);
        v16bf a1 = load_frag(Xs + (16 + idx) * KIN + kk + half * 8);
        const unsigned short* wb = Wih + (size_t)c * KIN + kk + half * 8;
        v16bf br = load_frag(wb);
        v16bf bz = load_frag(wb + (size_t)HH * KIN);
        v16bf bn = load_frag(wb + (size_t)2 * HH * KIN);
        axr0 = wmma_bf16(a0, br, axr0);  axr1 = wmma_bf16(a1, br, axr1);
        axz0 = wmma_bf16(a0, bz, axz0);  axz1 = wmma_bf16(a1, bz, axz1);
        axn0 = wmma_bf16(a0, bn, axn0);  axn1 = wmma_bf16(a1, bn, axn1);
    }
    #pragma unroll
    for (int kk = 0; kk < HH; kk += 32) {
        v16bf a0 = load_frag(Hs + idx * HH + kk + half * 8);
        v16bf a1 = load_frag(Hs + (16 + idx) * HH + kk + half * 8);
        const unsigned short* wb = Whh + (size_t)c * HH + kk + half * 8;
        v16bf br = load_frag(wb);
        v16bf bz = load_frag(wb + (size_t)HH * HH);
        v16bf bn = load_frag(wb + (size_t)2 * HH * HH);
        ahr0 = wmma_bf16(a0, br, ahr0);  ahr1 = wmma_bf16(a1, br, ahr1);
        ahz0 = wmma_bf16(a0, bz, ahz0);  ahz1 = wmma_bf16(a1, bz, ahz1);
        ahn0 = wmma_bf16(a0, bn, ahn0);  ahn1 = wmma_bf16(a1, bn, ahn1);
    }

    const float bxr = bih[c], bxz = bih[c + HH], bxn = bih[c + 2 * HH];
    const float bhr = bhh[c], bhz = bhh[c + HH], bhn = bhh[c + 2 * HH];

    const unsigned base = ((unsigned)node0 * HH + (unsigned)c + (unsigned)half * 8u * HH) * 4u;
    const char* const holdb = (const char*)hold;
    char* const houtb = (char*)hout;
    char* const hboutb = (char*)hbout;

    #pragma unroll
    for (int v = 0; v < 8; ++v) {                // sub-tile 0: rows v + 8*half
        const unsigned o = base + (unsigned)v * (HH * 4u);
        float r  = sigmoidf_(axr0[v] + bxr + ahr0[v] + bhr);
        float z  = sigmoidf_(axz0[v] + bxz + ahz0[v] + bhz);
        float ng = tanhf(axn0[v] + bxn + r * (ahn0[v] + bhn));
        float hn = (1.0f - z) * ng + z * *(const float*)(holdb + o);
        *(float*)(houtb + o) = hn;
        *(unsigned short*)(hboutb + (o >> 1)) = f2bf(hn);
    }
    if (node0 + 16 < NN) {                       // sub-tile 1 (guarded tail)
        #pragma unroll
        for (int v = 0; v < 8; ++v) {
            const unsigned o = base + (16u * HH + (unsigned)v * HH) * 4u;
            float r  = sigmoidf_(axr1[v] + bxr + ahr1[v] + bhr);
            float z  = sigmoidf_(axz1[v] + bxz + ahz1[v] + bhz);
            float ng = tanhf(axn1[v] + bxn + r * (ahn1[v] + bhn));
            float hn = (1.0f - z) * ng + z * *(const float*)(holdb + o);
            *(float*)(houtb + o) = hn;
            *(unsigned short*)(hboutb + (o >> 1)) = f2bf(hn);
        }
    }
}

// ---------------------------------------------------------------------------
extern "C" void kernel_launch(void* const* d_in, const int* in_sizes, int n_in,
                              void* d_out, int out_size, void* d_ws, size_t ws_size,
                              hipStream_t stream) {
    const float* x      = (const float*)d_in[0];
    const int*   edges  = (const int*)  d_in[1];
    const float* msg_W  = (const float*)d_in[2];
    const float* msg_b  = (const float*)d_in[3];
    const float* g0Wih  = (const float*)d_in[4];
    const float* g0Whh  = (const float*)d_in[5];
    const float* g0bih  = (const float*)d_in[6];
    const float* g0bhh  = (const float*)d_in[7];
    const float* g1Wih  = (const float*)d_in[8];
    const float* g1Whh  = (const float*)d_in[9];
    const float* g1bih  = (const float*)d_in[10];
    const float* g1bhh  = (const float*)d_in[11];

    // Workspace carve-out (~103 MB).
    char* ws = (char*)d_ws;
    size_t o = 0;
    auto carve = [&](size_t bytes) { char* p = ws + o; o += (bytes + 255) & ~(size_t)255; return p; };
    float*          h    = (float*)         carve((size_t)NN * HH * 4);
    float*          inc  = (float*)         carve((size_t)NN * HH * 4);
    unsigned short* hbuf = (unsigned short*)carve((size_t)NN * HH * 2);
    unsigned short* xin0 = (unsigned short*)carve((size_t)NN * HH * 2);
    unsigned short* xin1 = (unsigned short*)carve((size_t)NN * 2 * HH * 2);
    unsigned short* mWb  = (unsigned short*)carve((size_t)2 * TT * HH * HH * 2);
    unsigned short* w0ih = (unsigned short*)carve((size_t)384 * HH * 2);
    unsigned short* w0hh = (unsigned short*)carve((size_t)384 * HH * 2);
    unsigned short* w1ih = (unsigned short*)carve((size_t)384 * 2 * HH * 2);
    unsigned short* w1hh = (unsigned short*)carve((size_t)384 * HH * 2);

    auto cvt = [&](const float* src, unsigned short* dst, int n) {
        cvt_bf16<<<(n + 255) / 256, 256, 0, stream>>>(src, dst, n);
    };

    // Weight conversion (re-done every call: deterministic, graph-safe).
    cvt(msg_W, mWb, 2 * TT * HH * HH);
    cvt(g0Wih, w0ih, 384 * HH);
    cvt(g0Whh, w0hh, 384 * HH);
    cvt(g1Wih, w1ih, 384 * 2 * HH);
    cvt(g1Whh, w1hh, 384 * HH);

    const int nh = NN * HH;
    // xin1[:, 0:128] = bf16(x)  (constant across layer-1 timesteps)
    cvt_bf16_strided<<<(nh + 255) / 256, 256, 0, stream>>>(x, xin1, HH, 2 * HH, 0, nh);
    // h = x (f32 state); hbuf = bf16(x) (bf16 state, updated in-place by GRU)
    (void)hipMemcpyAsync(h, x, (size_t)NN * HH * 4, hipMemcpyDeviceToDevice, stream);
    cvt(x, hbuf, nh);

    const int gru_blocks = (NN + 31) / 32;
    for (int layer = 0; layer < 2; ++layer) {
        for (int step = 0; step < 3; ++step) {
            zero_f32<<<(nh + 255) / 256, 256, 0, stream>>>(inc, nh);

            dim3 mg(EE / EPB, TT);
            msg_kernel<<<mg, 256, 0, stream>>>(hbuf, edges,
                                               mWb + (size_t)layer * TT * HH * HH,
                                               msg_b + (size_t)layer * TT * HH,
                                               inc);

            const bool last = (layer == 1 && step == 2);
            float* hout = last ? (float*)d_out : h;
            if (layer == 0) {
                cvt(inc, xin0, nh);
                gru_kernel<HH><<<gru_blocks, 256, 0, stream>>>(
                    xin0, hbuf, h, w0ih, w0hh, g0bih, g0bhh, hout, hbuf);
            } else {
                cvt_bf16_strided<<<(nh + 255) / 256, 256, 0, stream>>>(
                    inc, xin1, HH, 2 * HH, HH, nh);
                gru_kernel<2 * HH><<<gru_blocks, 256, 0, stream>>>(
                    xin1, hbuf, h, w1ih, w1hh, g1bih, g1bhh, hout, hbuf);
            }
        }
    }
    (void)in_sizes; (void)n_in; (void)out_size; (void)ws_size;
}
